// HierAttention_59631325938146
// MI455X (gfx1250) — matrix operand order repaired and checked by reading, not compiled, not run on hardware
//
#include <hip/hip_runtime.h>
#include <hip/hip_bf16.h>
#include <math.h>

// Shapes from the reference
#define Bn   16
#define L1n  32
#define L2n  128
#define DECn 512
#define MEMn 512
#define TOPn 50

typedef __attribute__((ext_vector_type(16))) __bf16 v16bf;
typedef __attribute__((ext_vector_type(8)))  float  v8f;

// ---------------------------------------------------------------------------
// CDNA5 async global->LDS copy (16 bytes per lane), tracked by ASYNCcnt.
// LDS address is the LDS base plus the VDST VGPR; low 32 bits of a generic shared pointer
// give the in-allocation LDS byte offset.
// ---------------------------------------------------------------------------
__device__ __forceinline__ void async_cp_b128(unsigned lds_byte_off, const void* gptr) {
  asm volatile("global_load_async_to_lds_b128 %0, %1, off"
               :: "v"(lds_byte_off), "v"(gptr)
               : "memory");
}
__device__ __forceinline__ void wait_async_le12() {
  asm volatile("s_wait_asynccnt 12" ::: "memory");
}
__device__ __forceinline__ void wait_async_0() {
  asm volatile("s_wait_asynccnt 0" ::: "memory");
}

// ---------------------------------------------------------------------------
// K1: bias1[b,d] = Wd[d,:].dec[b,:] + Wt[d,:].topic[b,:]  (and bias2 w/ *2)
// ---------------------------------------------------------------------------
__global__ void k_prep_bias(const float* __restrict__ dec,
                            const float* __restrict__ topic,
                            const float* __restrict__ Wd,  const float* __restrict__ Wt,
                            const float* __restrict__ Wd2, const float* __restrict__ Wt2,
                            float* __restrict__ bias1, float* __restrict__ bias2) {
  int idx = blockIdx.x * 256 + threadIdx.x;      // 16*512 = 8192
  int b = idx >> 9, d = idx & 511;
  const float* xb = dec + b * DECn;
  const float* tb = topic + b * TOPn;
  float a1 = 0.f, a2 = 0.f;
  for (int k = 0; k < DECn; ++k) {
    float x = xb[k];
    a1 += Wd [d * DECn + k] * x;
    a2 += Wd2[d * DECn + k] * x;
  }
  for (int j = 0; j < TOPn; ++j) {
    float x = tb[j];
    a1 += Wt [d * TOPn + j] * x;
    a2 += Wt2[d * TOPn + j] * x;
  }
  bias1[idx] = a1;
  bias2[idx] = a2;
}

// ---------------------------------------------------------------------------
// K2: build bf16 B-matrix (Wm^T) in WMMA-fragment-linear order.
// Frag (kt,nt) covers K=kt*32..+32, N=nt*16..+16.  Per-lane 16 halves:
//   lanes 0-15 : K = kt*32 + h        (h=0..15), N = nt*16 + lane
//   lanes 16-31: K = kt*32 + 16 + h,            N = nt*16 + (lane-16)
// ---------------------------------------------------------------------------
__global__ void k_prep_bfrag(const float* __restrict__ Wm, __bf16* __restrict__ bfrag) {
  int fl = blockIdx.x * 256 + threadIdx.x;       // 512 frags * 32 lanes = 16384
  int f = fl >> 5, lane = fl & 31;
  int kt = f >> 5, nt = f & 31;
  int hi = lane >> 4;
  int n  = (nt << 4) + (lane & 15);
  int k0 = (kt << 5) + (hi << 4);
  const float* src = Wm + n * MEMn + k0;         // Wm is [DEC][MEM], B[k][n]=Wm[n][k]
  union { __bf16 h[16]; uint4 u[2]; } pk;
#pragma unroll
  for (int i = 0; i < 16; ++i) pk.h[i] = (__bf16)src[i];
  uint4* dst = reinterpret_cast<uint4*>(bfrag + (size_t)fl * 16);
  dst[0] = pk.u[0];
  dst[1] = pk.u[1];
}

// ---------------------------------------------------------------------------
// K3: Wm2^T (for coalesced doc-level dot products)
// ---------------------------------------------------------------------------
__global__ void k_prep_wm2t(const float* __restrict__ Wm2, float* __restrict__ wm2t) {
  int idx = blockIdx.x * 256 + threadIdx.x;      // 512*512
  int m = idx >> 9, d = idx & 511;
  wm2t[idx] = Wm2[d * MEMn + m];
}

// ---------------------------------------------------------------------------
// K4: doc-level attention (small: 0.27 GFLOP) -> doc_attn[b][q]
// ---------------------------------------------------------------------------
__global__ void k_doc(const float* __restrict__ doc_memory,
                      const int*   __restrict__ doc_mask,
                      const float* __restrict__ wm2t,
                      const float* __restrict__ Wv2,
                      const float* __restrict__ bias2,
                      float* __restrict__ doc_attn) {
  int b = blockIdx.x, t = threadIdx.x;
  __shared__ float docLDS[16 * 512];
  __shared__ float wsum[8];
  __shared__ float scoreLDS[32];
  float c[32];
  int d0 = t, d1 = t + 256;
  float bb0 = bias2[b * DECn + d0], bb1 = bias2[b * DECn + d1];
  float wv0 = Wv2[d0], wv1 = Wv2[d1];

  for (int half = 0; half < 2; ++half) {
    __syncthreads();
    for (int i = t; i < 16 * 512; i += 256)
      docLDS[i] = doc_memory[(b * L1n + half * 16) * MEMn + i];
    __syncthreads();
    float acc0[16], acc1[16];
#pragma unroll
    for (int q = 0; q < 16; ++q) { acc0[q] = 0.f; acc1[q] = 0.f; }
    for (int m = 0; m < MEMn; ++m) {
      float w0 = wm2t[m * DECn + d0];
      float w1 = wm2t[m * DECn + d1];
#pragma unroll
      for (int q = 0; q < 16; ++q) {
        float xm = docLDS[q * 512 + m];
        acc0[q] += w0 * xm;
        acc1[q] += w1 * xm;
      }
    }
#pragma unroll
    for (int q = 0; q < 16; ++q)
      c[half * 16 + q] = wv0 * tanhf(acc0[q] + bb0) + wv1 * tanhf(acc1[q] + bb1);
  }

  int lane = t & 31, widx = t >> 5;
  for (int q = 0; q < 32; ++q) {
    float v = c[q];
    v += __shfl_xor(v, 1);  v += __shfl_xor(v, 2);
    v += __shfl_xor(v, 4);  v += __shfl_xor(v, 8);
    v += __shfl_xor(v, 16);
    if (lane == 0) wsum[widx] = v;
    __syncthreads();
    if (t == 0) {
      float ss = 0.f;
#pragma unroll
      for (int k = 0; k < 8; ++k) ss += wsum[k];
      scoreLDS[q] = ss;
    }
    __syncthreads();
  }
  if (t < 32) {
    float sc = (doc_mask[b * L1n + t] == 0) ? -INFINITY : scoreLDS[t];
    float mx = sc;
    mx = fmaxf(mx, __shfl_xor(mx, 16)); mx = fmaxf(mx, __shfl_xor(mx, 8));
    mx = fmaxf(mx, __shfl_xor(mx, 4));  mx = fmaxf(mx, __shfl_xor(mx, 2));
    mx = fmaxf(mx, __shfl_xor(mx, 1));
    float e = expf(sc - mx), sm = e;
    sm += __shfl_xor(sm, 16); sm += __shfl_xor(sm, 8);
    sm += __shfl_xor(sm, 4);  sm += __shfl_xor(sm, 2);
    sm += __shfl_xor(sm, 1);
    doc_attn[b * L1n + t] = e / sm;
  }
}

// ---------------------------------------------------------------------------
// K5: fused word-level kernel.  One workgroup (8 waves) per (b,q).
//   GEMM 128x512x512 via v_wmma_f32_16x16x32_bf16 with double-buffered
//   async global->LDS staging (global_load_async_to_lds_b128 / ASYNCcnt),
//   fused tanh + Wv-reduce, softmax over w, rescale by doc_attn,
//   context partial from the fp32 tile (L2-hot).
// ---------------------------------------------------------------------------
__global__ __launch_bounds__(256)
void k_word(const float* __restrict__ word_memory,
            const int*   __restrict__ word_mask,
            const __bf16* __restrict__ bfrag,
            const float* __restrict__ Wv,
            const float* __restrict__ bias1,
            const float* __restrict__ doc_attn,
            float* __restrict__ out_resc,
            float* __restrict__ ctx_partial) {
  const int bq   = blockIdx.x;           // 512
  const int b    = bq >> 5;
  const int t    = threadIdx.x;
  const int lane = t & 31;
  const int wid  = t >> 5;

  // Double-buffered staging areas (async-to-LDS needs raw copies, so A is fp32)
  __shared__ __align__(16) float  aLDS[2][128 * 68];   // 64 m per slice, pad to 68
  __shared__ __align__(16) __bf16 bLDS[2][16 * 512];   // 16 frags (2 kt x 8 nt)
  __shared__ float  biasLDS[DECn];
  __shared__ float  wvLDS[DECn];
  __shared__ float  scoreLDS[L2n];
  __shared__ float  rescLDS[L2n];

  for (int i = t; i < DECn; i += 256) {
    biasLDS[i] = bias1[b * DECn + i];
    wvLDS[i]   = Wv[i];
  }
  __syncthreads();

  const float* wmB = word_memory + (size_t)bq * (L2n * MEMn);

  const unsigned aBase0 = (unsigned)(uintptr_t)&aLDS[0][0];
  const unsigned aBase1 = (unsigned)(uintptr_t)&aLDS[1][0];
  const unsigned bBase0 = (unsigned)(uintptr_t)&bLDS[0][0];
  const unsigned bBase1 = (unsigned)(uintptr_t)&bLDS[1][0];

  // Issue one K-slice stage (12 async b128 per thread: 8 for A, 4 for B).
  auto issue_stage = [&](int ng, int ks, int pb) {
    unsigned aB = pb ? aBase1 : aBase0;
    unsigned bB = pb ? bBase1 : bBase0;
#pragma unroll
    for (int it = 0; it < 8; ++it) {             // A: 128 rows x 64 floats
      int i  = t + it * 256;                     // 2048 16B chunks
      int w  = i >> 4, m4 = i & 15;
      unsigned loff = aB + (unsigned)((w * 68 + m4 * 4) * 4);
      async_cp_b128(loff, wmB + w * MEMn + ks * 64 + m4 * 4);
    }
#pragma unroll
    for (int it = 0; it < 4; ++it) {             // B: 16 frags x 1 KiB
      int i = t + it * 256;                      // 1024 16B chunks
      int f = i >> 6, r = i & 63;
      int ktl = f >> 3, j = f & 7;
      int ktg = ks * 2 + ktl, nt = ng * 8 + j;
      unsigned loff = bB + (unsigned)(f * 1024 + r * 16);
      async_cp_b128(loff, bfrag + (size_t)(ktg * 32 + nt) * 512 + r * 8);
    }
  };

  float s[8];
#pragma unroll
  for (int r = 0; r < 8; ++r) s[r] = 0.f;

  for (int ng = 0; ng < 4; ++ng) {       // groups of 8 N-tiles (128 columns)
    v8f C[8];
#pragma unroll
    for (int j = 0; j < 8; ++j) C[j] = (v8f){0.f,0.f,0.f,0.f,0.f,0.f,0.f,0.f};
    float wvr[8], bbr[8];
#pragma unroll
    for (int j = 0; j < 8; ++j) {
      int n = ng * 128 + j * 16 + (lane & 15);
      wvr[j] = wvLDS[n];
      bbr[j] = biasLDS[n];
    }

    issue_stage(ng, 0, 0);               // prologue prefetch
    int pb = 0;
    for (int ks = 0; ks < 8; ++ks) {     // K slices of 64
      if (ks < 7) {
        issue_stage(ng, ks + 1, pb ^ 1); // prefetch next slice
        wait_async_le12();               // previous group (12) complete, in order
      } else {
        wait_async_0();
      }
      __syncthreads();                   // data visible to all waves

      const float* aBuf = &aLDS[pb][0];
      const __bf16* bBuf = &bLDS[pb][0];
#pragma unroll
      for (int ktl = 0; ktl < 2; ++ktl) {
        // A fragment: M = lane&15 (row); fp32 -> bf16 at load time
        int w  = wid * 16 + (lane & 15);
        int hi = lane >> 4;
        const float* ap = aBuf + (w * 68 + ktl * 32 + hi * 16);
        float4 f0 = *reinterpret_cast<const float4*>(ap);
        float4 f1 = *reinterpret_cast<const float4*>(ap + 4);
        float4 f2 = *reinterpret_cast<const float4*>(ap + 8);
        float4 f3 = *reinterpret_cast<const float4*>(ap + 12);
        union { __bf16 h[16]; v16bf v; } A;
        A.h[0]  = (__bf16)f0.x; A.h[1]  = (__bf16)f0.y;
        A.h[2]  = (__bf16)f0.z; A.h[3]  = (__bf16)f0.w;
        A.h[4]  = (__bf16)f1.x; A.h[5]  = (__bf16)f1.y;
        A.h[6]  = (__bf16)f1.z; A.h[7]  = (__bf16)f1.w;
        A.h[8]  = (__bf16)f2.x; A.h[9]  = (__bf16)f2.y;
        A.h[10] = (__bf16)f2.z; A.h[11] = (__bf16)f2.w;
        A.h[12] = (__bf16)f3.x; A.h[13] = (__bf16)f3.y;
        A.h[14] = (__bf16)f3.z; A.h[15] = (__bf16)f3.w;
#pragma unroll
        for (int j = 0; j < 8; ++j) {
          union { uint4 u[2]; v16bf v; } Bf;
          int boff = ((ktl * 8 + j) * 32 + lane) * 16;
          Bf.u[0] = *reinterpret_cast<const uint4*>(&bBuf[boff]);
          Bf.u[1] = *reinterpret_cast<const uint4*>(&bBuf[boff + 8]);
          C[j] = __builtin_amdgcn_wmma_f32_16x16x32_bf16(
                     false, A.v, false, Bf.v, (short)0, C[j], false, false);
        }
      }
      __syncthreads();                   // done reading pb before it is re-staged
      pb ^= 1;
    }
    // fused epilogue for this column group: tanh + Wv-weighted accumulate
#pragma unroll
    for (int j = 0; j < 8; ++j) {
#pragma unroll
      for (int r = 0; r < 8; ++r)
        s[r] += wvr[j] * tanhf(C[j][r] + bbr[j]);
    }
  }

  // reduce scores across the 16 lanes of each half (columns)
#pragma unroll
  for (int r = 0; r < 8; ++r) {
    s[r] += __shfl_xor(s[r], 1);
    s[r] += __shfl_xor(s[r], 2);
    s[r] += __shfl_xor(s[r], 4);
    s[r] += __shfl_xor(s[r], 8);
  }
  if ((lane & 15) == 0) {
    int wbase = wid * 16 + (lane >> 4) * 8;   // lo half rows 0-7, hi half rows 8-15
#pragma unroll
    for (int r = 0; r < 8; ++r) scoreLDS[wbase + r] = s[r];
  }
  __syncthreads();

  // softmax over 128 w's (wave 0), rescale by doc_attn, emit rescaled
  if (wid == 0) {
    float docA = doc_attn[bq];
    float sc[4], ex[4];
    float mx = -INFINITY;
#pragma unroll
    for (int k = 0; k < 4; ++k) {
      int w = lane + k * 32;
      float v = scoreLDS[w];
      if (word_mask[bq * L2n + w] == 0) v = -INFINITY;
      sc[k] = v;
      mx = fmaxf(mx, v);
    }
    mx = fmaxf(mx, __shfl_xor(mx, 16)); mx = fmaxf(mx, __shfl_xor(mx, 8));
    mx = fmaxf(mx, __shfl_xor(mx, 4));  mx = fmaxf(mx, __shfl_xor(mx, 2));
    mx = fmaxf(mx, __shfl_xor(mx, 1));
    float sum = 0.f;
#pragma unroll
    for (int k = 0; k < 4; ++k) { ex[k] = expf(sc[k] - mx); sum += ex[k]; }
    sum += __shfl_xor(sum, 16); sum += __shfl_xor(sum, 8);
    sum += __shfl_xor(sum, 4);  sum += __shfl_xor(sum, 2);
    sum += __shfl_xor(sum, 1);
    float inv = 1.f / sum;
#pragma unroll
    for (int k = 0; k < 4; ++k) {
      int w = lane + k * 32;
      float rr = ex[k] * inv * docA;
      rescLDS[w] = rr;
      out_resc[bq * L2n + w] = rr;
    }
  }
  __syncthreads();

  // context partial (fp32, tile is L2-hot): ctx[m] = sum_w resc[w]*wm[w][m]
  for (int m = t; m < MEMn; m += 256) {
    float acc = 0.f;
    for (int w = 0; w < L2n; ++w)
      acc += rescLDS[w] * wmB[w * MEMn + m];
    ctx_partial[bq * MEMn + m] = acc;
  }
}

// ---------------------------------------------------------------------------
// K6: context[b][m] = sum_q ctx_partial[(b,q)][m]   (deterministic reduce)
// ---------------------------------------------------------------------------
__global__ void k_reduce(const float* __restrict__ ctx_partial, float* __restrict__ out_ctx) {
  int idx = blockIdx.x * 256 + threadIdx.x;      // 16*512
  int b = idx >> 9, m = idx & 511;
  float acc = 0.f;
  for (int q = 0; q < L1n; ++q)
    acc += ctx_partial[((b * L1n + q) << 9) + m];
  out_ctx[idx] = acc;
}

// ---------------------------------------------------------------------------
extern "C" void kernel_launch(void* const* d_in, const int* in_sizes, int n_in,
                              void* d_out, int out_size, void* d_ws, size_t ws_size,
                              hipStream_t stream) {
  const float* dec   = (const float*)d_in[0];
  const float* docm  = (const float*)d_in[1];
  const float* wordm = (const float*)d_in[2];
  const float* topic = (const float*)d_in[3];
  const int*   dmask = (const int*)d_in[4];
  const int*   wmask = (const int*)d_in[5];
  const float* Wv    = (const float*)d_in[6];
  const float* Wd    = (const float*)d_in[7];
  const float* Wt    = (const float*)d_in[8];
  const float* Wm    = (const float*)d_in[9];
  const float* Wv2   = (const float*)d_in[10];
  const float* Wd2   = (const float*)d_in[11];
  const float* Wt2   = (const float*)d_in[12];
  const float* Wm2   = (const float*)d_in[13];

  float* out_ctx  = (float*)d_out;               // [16,512]
  float* out_resc = (float*)d_out + Bn * MEMn;   // [16,32,128]

  char* ws = (char*)d_ws;
  __bf16* ws_bfrag = (__bf16*)(ws);              // 512 KiB
  float*  ws_bias1 = (float*)(ws + 524288);      // 32 KiB
  float*  ws_bias2 = (float*)(ws + 557056);      // 32 KiB
  float*  ws_wm2t  = (float*)(ws + 589824);      // 1 MiB
  float*  ws_docat = (float*)(ws + 1638400);     // 2 KiB
  float*  ws_ctxp  = (float*)(ws + 1640448);     // 1 MiB

  k_prep_bias <<<32,   256, 0, stream>>>(dec, topic, Wd, Wt, Wd2, Wt2, ws_bias1, ws_bias2);
  k_prep_bfrag<<<64,   256, 0, stream>>>(Wm, ws_bfrag);
  k_prep_wm2t <<<1024, 256, 0, stream>>>(Wm2, ws_wm2t);
  k_doc       <<<Bn,   256, 0, stream>>>(docm, dmask, ws_wm2t, Wv2, ws_bias2, ws_docat);
  k_word      <<<Bn * L1n, 256, 0, stream>>>(wordm, wmask, ws_bfrag, Wv, ws_bias1,
                                             ws_docat, out_resc, ws_ctxp);
  k_reduce    <<<32,   256, 0, stream>>>(ws_ctxp, out_ctx);
}
